// DalleBartDecoderFlax_43044162240901
// MI455X (gfx1250) — matrix operand order repaired and checked by reading, not compiled, not run on hardware
//
#include <hip/hip_runtime.h>
#include <hip/hip_bf16.h>
#include <math.h>

// ---- problem constants (match reference) ----
#define LNUM 12
#define BN   2
#define TIMG 256
#define TTXT 64
#define EN   1024
#define HN   16
#define CN   64
#define GN   2730
#define VN   16416
#define TB   256
#define KMAX 2730   // largest K for GEMV LDS staging

typedef __attribute__((ext_vector_type(16))) _Float16 v16h;
typedef __attribute__((ext_vector_type(8)))  float    v8f;

// ---------------------------------------------------------------------------
// embedding gather: out[b,e] = embed_tok[prev_token, e] + embed_pos[token_index, e]
__global__ void embed_kernel(const float* __restrict__ tok,
                             const float* __restrict__ pos,
                             const int* __restrict__ prev,
                             const int* __restrict__ tidx,
                             float* __restrict__ out) {
    int i = blockIdx.x * blockDim.x + threadIdx.x;
    if (i < BN * EN) {
        int e = i % EN;
        out[i] = tok[(size_t)(*prev) * EN + e] + pos[(size_t)(*tidx) * EN + e];
    }
}

// ---------------------------------------------------------------------------
// LayerNorm per row (block per row). scale/bias nullable. accumulate: out += LN(in)
__global__ void ln_kernel(const float* __restrict__ in,
                          const float* __restrict__ scale,
                          const float* __restrict__ bias,
                          float* __restrict__ out,
                          int n, int accumulate) {
    int row = blockIdx.x, t = threadIdx.x;
    const float* x = in + (size_t)row * n;
    float* y = out + (size_t)row * n;
    __shared__ float red[TB];
    float s = 0.f, ss = 0.f;
    for (int i = t; i < n; i += TB) { float v = x[i]; s += v; ss += v * v; }
    red[t] = s; __syncthreads();
    for (int o = TB / 2; o > 0; o >>= 1) { if (t < o) red[t] += red[t + o]; __syncthreads(); }
    s = red[0]; __syncthreads();
    red[t] = ss; __syncthreads();
    for (int o = TB / 2; o > 0; o >>= 1) { if (t < o) red[t] += red[t + o]; __syncthreads(); }
    ss = red[0];
    float mu  = s / n;
    float var = ss / n - mu * mu;
    float r   = rsqrtf(var + 1e-6f);
    for (int i = t; i < n; i += TB) {
        float v = (x[i] - mu) * r;
        if (scale) v *= scale[i];
        if (bias)  v += bias[i];
        if (accumulate) y[i] += v; else y[i] = v;
    }
}

// ---------------------------------------------------------------------------
__global__ void zero_kernel(float* __restrict__ p, int n) {
    int i = blockIdx.x * blockDim.x + threadIdx.x;
    if (i < n) p[i] = 0.f;
}

// ---------------------------------------------------------------------------
// Split-K GEMV (scalar columns) for 2 rows: y[b,n] += alpha * sum_k x[b,k]*W[k,n]
// Used only when N is not a multiple of 4 (fc0/fc1, N=2730).
__global__ void gemv2_kernel(const float* __restrict__ x,
                             const float* __restrict__ W,
                             float* __restrict__ y,
                             int K, int N, float alpha) {
    __shared__ float xs[2 * KMAX];
    int k0  = (int)(((long long)K * blockIdx.y) / gridDim.y);
    int k1  = (int)(((long long)K * (blockIdx.y + 1)) / gridDim.y);
    int len = k1 - k0;
    for (int i = threadIdx.x; i < 2 * len; i += TB) {
        int b = i / len, kk = i % len;
        xs[b * KMAX + kk] = x[(size_t)b * K + k0 + kk];
    }
    __syncthreads();
    int n = blockIdx.x * TB + threadIdx.x;
    if (n < N) {
        const float* wp = W + (size_t)k0 * N + n;
        float a0 = 0.f, a1 = 0.f;
#pragma unroll 4
        for (int kk = 0; kk < len; ++kk) {
            float w = wp[(size_t)kk * N];
            a0 += xs[kk] * w;
            a1 += xs[KMAX + kk] * w;
        }
        atomicAdd(&y[n], alpha * a0);
        atomicAdd(&y[N + n], alpha * a1);
    }
}

// ---------------------------------------------------------------------------
// Split-K GEMV, 4 columns/thread via float4 (b128) weight loads. Requires N%4==0.
__global__ void gemv2x4_kernel(const float* __restrict__ x,
                               const float* __restrict__ W,
                               float* __restrict__ y,
                               int K, int N, float alpha) {
    __shared__ float xs[2 * KMAX];
    int k0  = (int)(((long long)K * blockIdx.y) / gridDim.y);
    int k1  = (int)(((long long)K * (blockIdx.y + 1)) / gridDim.y);
    int len = k1 - k0;
    for (int i = threadIdx.x; i < 2 * len; i += TB) {
        int b = i / len, kk = i % len;
        xs[b * KMAX + kk] = x[(size_t)b * K + k0 + kk];
    }
    __syncthreads();
    int n4 = (blockIdx.x * TB + threadIdx.x) * 4;
    if (n4 < N) {
        const float4* wp = (const float4*)(W + (size_t)k0 * N + n4);
        size_t stride = (size_t)N >> 2;           // float4 row stride
        float4 a0 = make_float4(0.f, 0.f, 0.f, 0.f);
        float4 a1 = make_float4(0.f, 0.f, 0.f, 0.f);
#pragma unroll 4
        for (int kk = 0; kk < len; ++kk) {
            float4 w = wp[(size_t)kk * stride];
            float x0 = xs[kk], x1 = xs[KMAX + kk];
            a0.x += x0 * w.x; a0.y += x0 * w.y; a0.z += x0 * w.z; a0.w += x0 * w.w;
            a1.x += x1 * w.x; a1.y += x1 * w.y; a1.z += x1 * w.z; a1.w += x1 * w.w;
        }
        atomicAdd(&y[n4 + 0], alpha * a0.x);
        atomicAdd(&y[n4 + 1], alpha * a0.y);
        atomicAdd(&y[n4 + 2], alpha * a0.z);
        atomicAdd(&y[n4 + 3], alpha * a0.w);
        atomicAdd(&y[N + n4 + 0], alpha * a1.x);
        atomicAdd(&y[N + n4 + 1], alpha * a1.y);
        atomicAdd(&y[N + n4 + 2], alpha * a1.z);
        atomicAdd(&y[N + n4 + 3], alpha * a1.w);
    }
}

// ---------------------------------------------------------------------------
// scatter fresh K/V row into the forwarded cache at token_index
__global__ void kv_write_kernel(const float* __restrict__ kv,
                                float* __restrict__ cache,
                                int l, const int* __restrict__ tidx) {
    int i = blockIdx.x * blockDim.x + threadIdx.x;
    if (i < BN * EN) {
        int b = i / EN, e = i % EN;
        cache[(((size_t)l * BN + b) * TIMG + *tidx) * EN + e] = kv[i];
    }
}

// ---------------------------------------------------------------------------
// attention for one (b,h): scores over T keys, softmax, weighted V sum.
// mask: optional [B,T] bool (uint8). tidx: optional -> kmax = *tidx+1 (causal).
__global__ void attend_kernel(const float* __restrict__ q,   // [B, E] (h*C+c)
                              const float* __restrict__ Kc,  // [B, T, E]
                              const float* __restrict__ Vc,  // [B, T, E]
                              float* __restrict__ out,       // [B, E]
                              int T,
                              const unsigned char* __restrict__ mask,
                              const int* __restrict__ tidx) {
    int b = blockIdx.x / HN, h = blockIdx.x % HN, t = threadIdx.x;
    __shared__ float s[TIMG];
    __shared__ float red[TB];
    __shared__ float qs[CN];
    int kmax = tidx ? (*tidx + 1) : T;
    if (t < CN) qs[t] = q[(size_t)b * EN + h * CN + t];
    __syncthreads();
    for (int k = t; k < T; k += TB) {
        bool ok = (k < kmax) && (!mask || mask[(size_t)b * T + k]);
        float acc = -INFINITY;
        if (ok) {
            const float* kp = Kc + ((size_t)b * T + k) * EN + h * CN;
            acc = 0.f;
#pragma unroll 8
            for (int c = 0; c < CN; ++c) acc += qs[c] * kp[c];
        }
        s[k] = acc;
    }
    __syncthreads();
    float m = -INFINITY;
    for (int k = t; k < T; k += TB) m = fmaxf(m, s[k]);
    red[t] = m; __syncthreads();
    for (int o = TB / 2; o > 0; o >>= 1) { if (t < o) red[t] = fmaxf(red[t], red[t + o]); __syncthreads(); }
    m = red[0]; __syncthreads();
    float sum = 0.f;
    for (int k = t; k < T; k += TB) { float e = __expf(s[k] - m); s[k] = e; sum += e; }
    red[t] = sum; __syncthreads();
    for (int o = TB / 2; o > 0; o >>= 1) { if (t < o) red[t] += red[t + o]; __syncthreads(); }
    float inv = 1.f / red[0];
    __syncthreads();
    int tc = t & (CN - 1), part = t >> 6;       // 64 cols x 4 k-slices
    float acc = 0.f;
    for (int k = part; k < T; k += 4)
        acc += s[k] * Vc[((size_t)b * T + k) * EN + h * CN + tc];
    red[part * CN + tc] = acc; __syncthreads();
    if (part == 0)
        out[(size_t)b * EN + h * CN + tc] =
            (red[tc] + red[CN + tc] + red[2 * CN + tc] + red[3 * CN + tc]) * inv;
}

// ---------------------------------------------------------------------------
// out = gelu_exact(h0) * h1
__global__ void gelu_mul_kernel(const float* __restrict__ h0,
                                const float* __restrict__ h1,
                                float* __restrict__ out, int n) {
    int i = blockIdx.x * blockDim.x + threadIdx.x;
    if (i < n) {
        float v = h0[i];
        float g = 0.5f * v * (1.f + erff(v * 0.70710678118654752f));
        out[i] = g * h1[i];
    }
}

// ---------------------------------------------------------------------------
// WMMA GEMM: Y[M,N] = A[M,K] * W[K,N], fp32 in memory, fp16 WMMA compute.
// One wave per 16x32 output strip: A fragment shared across two N-tiles,
// two independent v_wmma accumulators per K-step (no D->A/B RAW hazard).
// A fragment (16x32, contiguous k order matching B): lane(m = L&15, hi = L>>4):
//   half j: k = hi*16 + j
// B fragment (32x16): lane(n = L&15, hi = L>>4): half j -> k = hi*16 + j
// D: lane n = L&15; VGPR r holds row M = r + (L>>4)*8
__global__ void wmma_gemm_f16_kernel(const float* __restrict__ A,
                                     const float* __restrict__ W,
                                     float* __restrict__ Y,
                                     int M, int N, int K) {
    int lane = threadIdx.x & 31;
    int m0 = blockIdx.x * 16, n0 = blockIdx.y * 32;
    int mr = lane & 15, hi = lane >> 4;
    v8f acc0 = {}, acc1 = {};
    for (int kk = 0; kk < K; kk += 32) {
        if (kk + 32 < K)  // stream-ahead hint -> global_prefetch_b8
            __builtin_prefetch(&W[(size_t)(kk + 32 + hi * 16) * N + n0 + mr], 0, 1);
        v16h a, b0, b1;
        const float* ap = A + (size_t)(m0 + mr) * K + kk + hi * 16;
#pragma unroll
        for (int g = 0; g < 8; ++g) {
            int base = 2 * g;
            a[2 * g + 0] = (_Float16)ap[base + 0];
            a[2 * g + 1] = (_Float16)ap[base + 1];
        }
        const float* bp = W + (size_t)(kk + hi * 16) * N + n0 + mr;
#pragma unroll
        for (int j = 0; j < 16; ++j) {
            b0[j] = (_Float16)bp[(size_t)j * N];
            b1[j] = (_Float16)bp[(size_t)j * N + 16];
        }
        acc0 = __builtin_amdgcn_wmma_f32_16x16x32_f16(false, a, false, b0,
                                                      (short)0, acc0, false, false);
        acc1 = __builtin_amdgcn_wmma_f32_16x16x32_f16(false, a, false, b1,
                                                      (short)0, acc1, false, false);
    }
    int n = lane & 15;
#pragma unroll
    for (int r = 0; r < 8; ++r) {
        Y[(size_t)(m0 + r + hi * 8) * N + n0 + n]      = acc0[r];
        Y[(size_t)(m0 + r + hi * 8) * N + n0 + 16 + n] = acc1[r];
    }
}

// ---------------------------------------------------------------------------
extern "C" void kernel_launch(void* const* d_in, const int* in_sizes, int n_in,
                              void* d_out, int out_size, void* d_ws, size_t ws_size,
                              hipStream_t stream) {
    (void)in_sizes; (void)n_in; (void)out_size; (void)ws_size;
    const float* enc      = (const float*)d_in[0];
    const float* keys_in  = (const float*)d_in[1];
    const float* vals_in  = (const float*)d_in[2];
    const unsigned char* amask = (const unsigned char*)d_in[3];
    const int*   prev     = (const int*)d_in[4];
    const int*   tidx     = (const int*)d_in[5];
    const float* etok     = (const float*)d_in[6];
    const float* epos     = (const float*)d_in[7];
    const float* ln_emb_s = (const float*)d_in[8];
    const float* ln_emb_b = (const float*)d_in[9];
    const float* final_b  = (const float*)d_in[10];
    const float* lm_head  = (const float*)d_in[11];
    const float* pre_sa_b = (const float*)d_in[12];
    const float* sa_q_w   = (const float*)d_in[13];
    const float* sa_k_w   = (const float*)d_in[14];
    const float* sa_v_w   = (const float*)d_in[15];
    const float* sa_o_w   = (const float*)d_in[16];
    const float* sa_ln_s  = (const float*)d_in[17];
    const float* sa_ln_b  = (const float*)d_in[18];
    const float* pre_ca_b = (const float*)d_in[19];
    const float* ca_q_w   = (const float*)d_in[20];
    const float* ca_k_w   = (const float*)d_in[21];
    const float* ca_v_w   = (const float*)d_in[22];
    const float* ca_o_w   = (const float*)d_in[23];
    const float* ca_ln_s  = (const float*)d_in[24];
    const float* ca_ln_b  = (const float*)d_in[25];
    const float* g0b      = (const float*)d_in[26];
    const float* f0w      = (const float*)d_in[27];
    const float* f1w      = (const float*)d_in[28];
    const float* g1b      = (const float*)d_in[29];
    const float* f2w      = (const float*)d_in[30];

    // output layout: [logits B*V | keys L*B*TIMG*E | values L*B*TIMG*E]
    float* outf     = (float*)d_out;
    float* logits   = outf;
    float* keys_out = outf + (size_t)BN * VN;
    float* vals_out = keys_out + (size_t)LNUM * BN * TIMG * EN;
    const size_t cache_bytes = (size_t)LNUM * BN * TIMG * EN * sizeof(float);
    hipMemcpyAsync(keys_out, keys_in, cache_bytes, hipMemcpyDeviceToDevice, stream);
    hipMemcpyAsync(vals_out, vals_in, cache_bytes, hipMemcpyDeviceToDevice, stream);

    // workspace layout (floats)
    float* w   = (float*)d_ws;
    float* ds  = w;                 // [B,E]
    float* xb  = w + 2048;          // [B,E]
    float* kvt = w + 4096;          // [B,E]
    float* qb  = w + 6144;          // [B,E]
    float* ab  = w + 8192;          // [B,E]
    float* pb  = w + 10240;         // [B,E]
    float* h0  = w + 12288;         // [B,G] (5460 used)
    float* h1  = w + 17792;         // [B,G]
    float* ek  = w + 23296;         // [B*TTXT, E]
    float* ev  = w + 154368;        // [B*TTXT, E]

    const float qscale = 1.f / 8.f;   // 1/sqrt(C)
    // vectorized GEMV grids: grid.x tiles N/4 by TB threads, grid.y = split-K depth
    dim3 gEE(((EN / 4) + TB - 1) / TB, 96);     // E->E : 96 blocks in flight
    dim3 gGE(((EN / 4) + TB - 1) / TB, 64);     // G->E (K=2730)
    dim3 gEV(((VN / 4) + TB - 1) / TB, 8);      // lm_head: 17 x 8 = 136 blocks
    dim3 gEG((GN + TB - 1) / TB, 16);           // E->G scalar (N=2730 not /4)
    int  nBE = (BN * EN + TB - 1) / TB;
    int  nBG = (BN * GN + TB - 1) / TB;
    dim3 wg(128 / 16, EN / 32);                 // wmma 16x32 strips for [128,1024,1024]

    // --- embedding + embed LN ---
    embed_kernel<<<nBE, TB, 0, stream>>>(etok, epos, prev, tidx, ds);
    ln_kernel<<<BN, TB, 0, stream>>>(ds, ln_emb_s, ln_emb_b, ds, EN, 0);

    for (int l = 0; l < LNUM; ++l) {
        const float* qw  = sa_q_w + (size_t)l * EN * EN;
        const float* kw  = sa_k_w + (size_t)l * EN * EN;
        const float* vw  = sa_v_w + (size_t)l * EN * EN;
        const float* ow  = sa_o_w + (size_t)l * EN * EN;
        const float* cqw = ca_q_w + (size_t)l * EN * EN;
        const float* ckw = ca_k_w + (size_t)l * EN * EN;
        const float* cvw = ca_v_w + (size_t)l * EN * EN;
        const float* cow = ca_o_w + (size_t)l * EN * EN;
        const float* f0  = f0w + (size_t)l * EN * GN;
        const float* f1  = f1w + (size_t)l * EN * GN;
        const float* f2  = f2w + (size_t)l * GN * EN;
        float* kcache = keys_out + (size_t)l * BN * TIMG * EN;
        float* vcache = vals_out + (size_t)l * BN * TIMG * EN;

        // --- self attention ---
        ln_kernel<<<BN, TB, 0, stream>>>(ds, nullptr, pre_sa_b + (size_t)l * EN, xb, EN, 0);
        zero_kernel<<<nBE, TB, 0, stream>>>(kvt, BN * EN);
        gemv2x4_kernel<<<gEE, TB, 0, stream>>>(xb, kw, kvt, EN, EN, 1.f);
        kv_write_kernel<<<nBE, TB, 0, stream>>>(kvt, keys_out, l, tidx);
        zero_kernel<<<nBE, TB, 0, stream>>>(kvt, BN * EN);
        gemv2x4_kernel<<<gEE, TB, 0, stream>>>(xb, vw, kvt, EN, EN, 1.f);
        kv_write_kernel<<<nBE, TB, 0, stream>>>(kvt, vals_out, l, tidx);
        zero_kernel<<<nBE, TB, 0, stream>>>(qb, BN * EN);
        gemv2x4_kernel<<<gEE, TB, 0, stream>>>(xb, qw, qb, EN, EN, qscale);
        attend_kernel<<<BN * HN, TB, 0, stream>>>(qb, kcache, vcache, ab, TIMG, nullptr, tidx);
        zero_kernel<<<nBE, TB, 0, stream>>>(pb, BN * EN);
        gemv2x4_kernel<<<gEE, TB, 0, stream>>>(ab, ow, pb, EN, EN, 1.f);
        ln_kernel<<<BN, TB, 0, stream>>>(pb, sa_ln_s + (size_t)l * EN, sa_ln_b + (size_t)l * EN, ds, EN, 1);

        // --- cross attention (encoder K/V projections via WMMA) ---
        ln_kernel<<<BN, TB, 0, stream>>>(ds, nullptr, pre_ca_b + (size_t)l * EN, xb, EN, 0);
        wmma_gemm_f16_kernel<<<wg, 32, 0, stream>>>(enc, ckw, ek, BN * TTXT, EN, EN);
        wmma_gemm_f16_kernel<<<wg, 32, 0, stream>>>(enc, cvw, ev, BN * TTXT, EN, EN);
        zero_kernel<<<nBE, TB, 0, stream>>>(qb, BN * EN);
        gemv2x4_kernel<<<gEE, TB, 0, stream>>>(xb, cqw, qb, EN, EN, qscale);
        attend_kernel<<<BN * HN, TB, 0, stream>>>(qb, ek, ev, ab, TTXT, amask, nullptr);
        zero_kernel<<<nBE, TB, 0, stream>>>(pb, BN * EN);
        gemv2x4_kernel<<<gEE, TB, 0, stream>>>(ab, cow, pb, EN, EN, 1.f);
        ln_kernel<<<BN, TB, 0, stream>>>(pb, ca_ln_s + (size_t)l * EN, ca_ln_b + (size_t)l * EN, ds, EN, 1);

        // --- GLU FFN ---
        ln_kernel<<<BN, TB, 0, stream>>>(ds, nullptr, g0b + (size_t)l * EN, xb, EN, 0);
        zero_kernel<<<nBG, TB, 0, stream>>>(h0, BN * GN);
        gemv2_kernel<<<gEG, TB, 0, stream>>>(xb, f0, h0, EN, GN, 1.f);
        zero_kernel<<<nBG, TB, 0, stream>>>(h1, BN * GN);
        gemv2_kernel<<<gEG, TB, 0, stream>>>(xb, f1, h1, EN, GN, 1.f);
        gelu_mul_kernel<<<nBG, TB, 0, stream>>>(h0, h1, h0, BN * GN);
        ln_kernel<<<BN, TB, 0, stream>>>(h0, nullptr, g1b + (size_t)l * GN, h1, GN, 0);
        gemv2x4_kernel<<<gGE, TB, 0, stream>>>(h1, f2, ds, GN, EN, 1.f);  // residual accumulate
    }

    // --- final LN + lm_head ---
    ln_kernel<<<BN, TB, 0, stream>>>(ds, nullptr, final_b, xb, EN, 0);
    zero_kernel<<<(BN * VN + TB - 1) / TB, TB, 0, stream>>>(logits, BN * VN);
    gemv2x4_kernel<<<gEV, TB, 0, stream>>>(xb, lm_head, logits, EN, VN, 1.f);
}
